// SemanticAlignmentModule_90469191123550
// MI455X (gfx1250) — matrix operand order repaired and checked by reading, not compiled, not run on hardware
//
#include <hip/hip_runtime.h>
#include <hip/hip_bf16.h>

// ---------------------------------------------------------------------------
// Problem constants (from reference): B=8, HS=WS=64, C=512, T=77, TD=512,
// HH=8 heads, DH=64, TOPM=3, FFN hidden = 2048. N = 4096, rows = B*N = 32768.
// ---------------------------------------------------------------------------
#define B_    8
#define CCH   512
#define TT    77
#define TDD   512
#define NHH   8
#define DHH   64
#define NROW  32768
#define FFN   2048

typedef __attribute__((ext_vector_type(16))) __bf16 v16bf;
typedef __attribute__((ext_vector_type(8)))  float  v8f;

union BF16Frag { v16bf v; unsigned short u[16]; uint4 q[2]; };

__device__ __forceinline__ unsigned short f2bf(float f) {
  unsigned int u = __float_as_uint(f);
  u += 0x7FFFu + ((u >> 16) & 1u);
  return (unsigned short)(u >> 16);
}

__device__ __forceinline__ float wave_sum(float x) {
  #pragma unroll
  for (int off = 16; off > 0; off >>= 1) x += __shfl_xor(x, off, 32);
  return x;
}
__device__ __forceinline__ float wave_max(float x) {
  #pragma unroll
  for (int off = 16; off > 0; off >>= 1) x = fmaxf(x, __shfl_xor(x, off, 32));
  return x;
}

// ---------------------------------------------------------------------------
// f32 -> bf16 conversion (weights)
// ---------------------------------------------------------------------------
__global__ void cvt_bf16_kernel(const float* __restrict__ s,
                                unsigned short* __restrict__ d, int n) {
  int i = blockIdx.x * blockDim.x + threadIdx.x;
  if (i < n) d[i] = f2bf(s[i]);
}

// ---------------------------------------------------------------------------
// LN1: per-row (C=512) LayerNorm; one wave per row, 16 elems per lane.
// Writes f32 (residual path) and bf16 (WMMA operand).
// ---------------------------------------------------------------------------
__global__ __launch_bounds__(256)
void ln1_kernel(const float* __restrict__ x, const float* __restrict__ w,
                const float* __restrict__ b, float* __restrict__ yf,
                unsigned short* __restrict__ ybf) {
  int wv = threadIdx.x >> 5, lane = threadIdx.x & 31;
  int row = blockIdx.x * 8 + wv;
  size_t base = (size_t)row * CCH + lane * 16;
  float vals[16]; float s = 0.f, s2 = 0.f;
  #pragma unroll
  for (int i = 0; i < 16; ++i) {
    float t = x[base + i]; vals[i] = t; s += t; s2 += t * t;
  }
  s = wave_sum(s); s2 = wave_sum(s2);
  float mu = s * (1.f / CCH);
  float var = s2 * (1.f / CCH) - mu * mu;
  float rstd = rsqrtf(var + 1e-5f);
  #pragma unroll
  for (int i = 0; i < 16; ++i) {
    int c = lane * 16 + i;
    float t = (vals[i] - mu) * rstd * w[c] + b[c];
    yf[base + i] = t;
    ybf[base + i] = f2bf(t);
  }
}

// ---------------------------------------------------------------------------
// K/V projections: only 616 rows -> plain VALU GEMV per block (one text row).
// ---------------------------------------------------------------------------
__global__ __launch_bounds__(256)
void kv_proj_kernel(const float* __restrict__ text,
                    const float* __restrict__ wk, const float* __restrict__ bk,
                    const float* __restrict__ wvp, const float* __restrict__ bv,
                    float* __restrict__ kf, float* __restrict__ vf) {
  __shared__ float st[TDD];
  int row = blockIdx.x;
  int tid = threadIdx.x;
  st[tid]       = text[(size_t)row * TDD + tid];
  st[tid + 256] = text[(size_t)row * TDD + tid + 256];
  __syncthreads();
  for (int cc = 0; cc < 2; ++cc) {
    int c = tid + cc * 256;
    float ak = 0.f, av = 0.f;
    for (int d = 0; d < TDD; ++d) {
      float t = st[d];
      ak += t * wk[(size_t)d * CCH + c];
      av += t * wvp[(size_t)d * CCH + c];
    }
    kf[(size_t)row * CCH + c] = ak + bk[c];
    vf[(size_t)row * CCH + c] = av + bv[c];
  }
}

__global__ __launch_bounds__(256)
void padmask_kernel(const float* __restrict__ text, float* __restrict__ pm, int rows) {
  int wv = threadIdx.x >> 5, lane = threadIdx.x & 31;
  int row = blockIdx.x * 8 + wv;
  if (row >= rows) return;
  float s = 0.f;
  for (int i = lane; i < TDD; i += 32) s += fabsf(text[(size_t)row * TDD + i]);
  s = wave_sum(s);
  if (lane == 0) pm[row] = (s <= 1e-6f) ? 1.f : 0.f;
}

__global__ __launch_bounds__(256)
void l2norm_kernel(float* __restrict__ x, int rows) {
  int wv = threadIdx.x >> 5, lane = threadIdx.x & 31;
  int row = blockIdx.x * 8 + wv;
  if (row >= rows) return;
  size_t base = (size_t)row * CCH;
  float s = 0.f;
  for (int i = lane; i < CCH; i += 32) { float t = x[base + i]; s += t * t; }
  s = wave_sum(s);
  float inv = 1.f / fmaxf(sqrtf(s), 1e-6f);
  for (int i = lane; i < CCH; i += 32) x[base + i] *= inv;
}

// ---------------------------------------------------------------------------
// bf16 WMMA GEMM: C[M,N] = A[M,K] * B[K,N] (+ epilogue).
// Block: 256 threads (8 waves), tile 128(M) x 64(N), BK=32.
// Waves laid out 4(M) x 2(N); each wave owns a 32x32 sub-tile =
// 2x2 v_wmma_f32_16x16x32_bf16 accumulators.
// A staged [m][k] (stride 40), B staged TRANSPOSED [n][k] (stride 40) so that
// every operand fragment is two contiguous 16-byte LDS reads (ds_load_b128):
//   A frag lane(l):  m = wm+16i+(l&15), u[0..15] = K {8h..8h+7, 16+8h..16+8h+7}
//   B frag lane(l):  n = wn+16j+(l&15), u[0..15] = K {8h..8h+7, 16+8h..16+8h+7}
// (h = l>>4), matching the ISA operand element order exactly.
// EPI: 0 = f32 out; 1 = f32 * rowscale (conf); 2 = sigmoid * clip(geo,.3,1);
//      3 = exact GELU -> bf16 out; 4 = + residual -> f32 out.
// ---------------------------------------------------------------------------
template <int EPI>
__global__ __launch_bounds__(256)
void gemm_bf16_kernel(const unsigned short* __restrict__ A,
                      const unsigned short* __restrict__ Bw,
                      const float* __restrict__ bias,
                      const float* __restrict__ rowscale,
                      const float* __restrict__ residual,
                      float* __restrict__ Cf,
                      unsigned short* __restrict__ Cbf,
                      int M, int N, int K) {
  __shared__ unsigned short As[128 * 40];  // [m][k], 80B row stride (16B mult)
  __shared__ unsigned short Bs[64 * 40];   // [n][k] transposed, 80B row stride
  int tid = threadIdx.x;
  int bm = blockIdx.x * 128;
  int bn = blockIdx.y * 64;
  int wv = tid >> 5, lane = tid & 31;
  int wm = (wv & 3) * 32;
  int wn = (wv >> 2) * 32;
  int lh = lane >> 4;       // lane half
  int l15 = lane & 15;

  v8f acc[2][2];
  #pragma unroll
  for (int i = 0; i < 2; ++i)
    #pragma unroll
    for (int j = 0; j < 2; ++j)
      #pragma unroll
      for (int e = 0; e < 8; ++e) acc[i][j][e] = 0.f;

  int ar = tid >> 1, ac = (tid & 1) * 16;  // A stage: 16 bf16 per thread
  int br = tid >> 3, bc = (tid & 7) * 8;   // B stage: 8 bf16 (1 k, 8 n)

  for (int k0 = 0; k0 < K; k0 += 32) {
    const uint4* ga = (const uint4*)(A + (size_t)(bm + ar) * K + k0 + ac);
    uint4 a0 = ga[0], a1 = ga[1];
    *(uint4*)(&As[ar * 40 + ac])     = a0;
    *(uint4*)(&As[ar * 40 + ac + 8]) = a1;
    union { uint4 q; unsigned short us[8]; } bld;
    bld.q = *(const uint4*)(Bw + (size_t)(k0 + br) * N + bn + bc);
    #pragma unroll
    for (int i2 = 0; i2 < 8; ++i2)      // transpose scatter: Bs[n][k]
      Bs[(bc + i2) * 40 + br] = bld.us[i2];
    if (k0 + 32 < K) {  // global_prefetch_b8 for next K tile
      __builtin_prefetch(A + (size_t)(bm + ar) * K + k0 + 32 + ac, 0, 1);
      __builtin_prefetch(Bw + (size_t)(k0 + 32 + br) * N + bn + bc, 0, 1);
    }
    __syncthreads();

    BF16Frag fa[2], fb[2];
    const unsigned short* a0p = &As[(wm + l15) * 40 + 8 * lh];
    fa[0].q[0] = *(const uint4*)(a0p);
    fa[0].q[1] = *(const uint4*)(a0p + 16);
    const unsigned short* a1p = &As[(wm + 16 + l15) * 40 + 8 * lh];
    fa[1].q[0] = *(const uint4*)(a1p);
    fa[1].q[1] = *(const uint4*)(a1p + 16);
    const unsigned short* b0p = &Bs[(wn + l15) * 40 + 8 * lh];
    fb[0].q[0] = *(const uint4*)(b0p);
    fb[0].q[1] = *(const uint4*)(b0p + 16);
    const unsigned short* b1p = &Bs[(wn + 16 + l15) * 40 + 8 * lh];
    fb[1].q[0] = *(const uint4*)(b1p);
    fb[1].q[1] = *(const uint4*)(b1p + 16);

    #pragma unroll
    for (int i = 0; i < 2; ++i)
      #pragma unroll
      for (int j = 0; j < 2; ++j)
        acc[i][j] = __builtin_amdgcn_wmma_f32_16x16x32_bf16(
            false, fa[i].v, false, fb[j].v, (short)0, acc[i][j], false, false);
    __syncthreads();
  }

  // Epilogue. C/D f32 layout: col = lane&15, row = v + 8*(lane>=16).
  #pragma unroll
  for (int i = 0; i < 2; ++i) {
    #pragma unroll
    for (int j = 0; j < 2; ++j) {
      int col = bn + wn + j * 16 + l15;
      #pragma unroll
      for (int e = 0; e < 8; ++e) {
        int row = bm + wm + i * 16 + e + 8 * lh;
        float x = acc[i][j][e] + bias[col];
        size_t idx = (size_t)row * N + col;
        if (EPI == 0) {
          Cf[idx] = x;
        } else if (EPI == 1) {
          Cf[idx] = x * rowscale[row];
        } else if (EPI == 2) {
          float g = fminf(fmaxf(rowscale[row], 0.3f), 1.f);
          Cf[idx] = g / (1.f + __expf(-x));
        } else if (EPI == 3) {
          float ge = 0.5f * x * (1.f + erff(x * 0.70710678118f));
          Cbf[idx] = f2bf(ge);
        } else {
          Cf[idx] = x + residual[idx];
        }
      }
    }
  }
}

// ---------------------------------------------------------------------------
// Attention: one wave per query row. T=77, top-3 sparsified softmax with geo
// log-bias and pad masking, entropy-based confidence, attn @ V -> bf16.
// ---------------------------------------------------------------------------
__global__ __launch_bounds__(128)
void attn_kernel(const float* __restrict__ q, const float* __restrict__ k,
                 const float* __restrict__ vmat, const float* __restrict__ pm,
                 const float* __restrict__ geo, const float* __restrict__ lsp,
                 unsigned short* __restrict__ alignedBf,
                 float* __restrict__ conf_scale) {
  __shared__ float s_q[4][64];
  __shared__ float s_attn[4][80];
  int wv = threadIdx.x >> 5, lane = threadIdx.x & 31;
  int row = blockIdx.x * 4 + wv;
  int b = row >> 12;  // N = 4096 rows per batch
  float ls = fminf(fmaxf(lsp[0], -2.f), 2.f);
  float scale = __expf(ls) * 0.125f;  // 1/sqrt(DH=64)
  float gg = fminf(fmaxf(geo[row], 0.3f), 1.f);
  float logg = __logf(fmaxf(gg, 1e-6f));
  const float NEG = -3.0e38f;
  float conf_sum = 0.f;

  for (int h = 0; h < NHH; ++h) {
    s_q[wv][lane]      = q[(size_t)row * CCH + h * DHH + lane];
    s_q[wv][lane + 32] = q[(size_t)row * CCH + h * DHH + lane + 32];
    __syncthreads();

    float sims[3];
    #pragma unroll
    for (int j = 0; j < 3; ++j) {
      int t = lane + j * 32;
      float s = NEG;
      if (t < TT && pm[b * TT + t] == 0.f) {
        const float* kr = k + ((size_t)(b * TT + t)) * CCH + h * DHH;
        float a0 = 0.f;
        #pragma unroll 8
        for (int d = 0; d < DHH; ++d) a0 += s_q[wv][d] * kr[d];
        s = a0 * scale + logg;
      }
      sims[j] = s;
    }
    // local sorted-3 (desc)
    float ta = sims[0], tb = sims[1], tc = sims[2], tmp;
    if (tb > ta) { tmp = ta; ta = tb; tb = tmp; }
    if (tc > ta) { tmp = ta; ta = tc; tc = tmp; }
    if (tc > tb) { tmp = tb; tb = tc; tc = tmp; }
    auto ins = [&](float x) {
      if (x > ta) { tc = tb; tb = ta; ta = x; }
      else if (x > tb) { tc = tb; tb = x; }
      else if (x > tc) { tc = x; }
    };
    #pragma unroll
    for (int off = 16; off > 0; off >>= 1) {
      float oa = __shfl_xor(ta, off, 32);
      float ob = __shfl_xor(tb, off, 32);
      float oc = __shfl_xor(tc, off, 32);
      ins(oa); ins(ob); ins(oc);
    }
    float thr = tc, mx = ta;
    bool anyvalid = (mx > -1e37f);
    float ev[3]; float esum = 0.f;
    #pragma unroll
    for (int j = 0; j < 3; ++j) {
      float e = (anyvalid && sims[j] >= thr) ? __expf(sims[j] - mx) : 0.f;
      ev[j] = e; esum += e;
    }
    float tot = wave_sum(esum);
    float inv = 1.f / fmaxf(tot, 1e-6f);
    float attnj[3]; float maxp_loc = 0.f, cnt_loc = 0.f, psum_loc = 0.f;
    #pragma unroll
    for (int j = 0; j < 3; ++j) {
      float a = ev[j] * inv;
      attnj[j] = a;
      int t = lane + j * 32;
      if (t < TT) {
        if (a > 0.f) cnt_loc += 1.f;
        float p = fmaxf(a, 1e-8f);
        psum_loc += p * __logf(p);
      }
      if (a > maxp_loc) maxp_loc = a;
    }
    s_attn[wv][lane]      = attnj[0];
    s_attn[wv][lane + 32] = attnj[1];
    if (lane + 64 < 80) s_attn[wv][lane + 64] = attnj[2];
    float maxp = wave_max(maxp_loc);
    float cnt  = wave_sum(cnt_loc);
    float psum = wave_sum(psum_loc);
    float teff = fmaxf(cnt, 2.f);
    float ent = fmaxf(-psum / __logf(teff), 0.f);
    conf_sum += fminf(fmaxf(maxp * (1.f - ent), 0.f), 1.f);
    __syncthreads();

    float a0 = 0.f, a1 = 0.f;
    for (int t = 0; t < TT; ++t) {
      float a = s_attn[wv][t];
      if (a != 0.f) {
        const float* vr = vmat + ((size_t)(b * TT + t)) * CCH + h * DHH;
        a0 += a * vr[lane];
        a1 += a * vr[lane + 32];
      }
    }
    alignedBf[(size_t)row * CCH + h * DHH + lane]      = f2bf(a0);
    alignedBf[(size_t)row * CCH + h * DHH + lane + 32] = f2bf(a1);
    __syncthreads();
  }
  if (lane == 0) {
    float conf = fminf(fmaxf(conf_sum * (1.f / NHH), 0.f), 1.f);
    conf_scale[row] = 0.35f + 0.65f * conf;
  }
}

// ---------------------------------------------------------------------------
// y = x + alpha*gate*o (in place over x), then LN2 -> bf16 FFN operand.
// ---------------------------------------------------------------------------
__global__ __launch_bounds__(256)
void combine_ln2_kernel(float* __restrict__ y, const float* __restrict__ o,
                        const float* __restrict__ gate,
                        const float* __restrict__ alpha_p,
                        const float* __restrict__ w, const float* __restrict__ bb,
                        unsigned short* __restrict__ hbf) {
  int wv = threadIdx.x >> 5, lane = threadIdx.x & 31;
  int row = blockIdx.x * 8 + wv;
  size_t base = (size_t)row * CCH + lane * 16;
  float alpha = alpha_p[0];
  float vals[16]; float s = 0.f, s2 = 0.f;
  #pragma unroll
  for (int i = 0; i < 16; ++i) {
    float t = y[base + i] + alpha * gate[base + i] * o[base + i];
    vals[i] = t; s += t; s2 += t * t;
  }
  s = wave_sum(s); s2 = wave_sum(s2);
  float mu = s * (1.f / CCH);
  float var = s2 * (1.f / CCH) - mu * mu;
  float rstd = rsqrtf(var + 1e-5f);
  #pragma unroll
  for (int i = 0; i < 16; ++i) {
    int c = lane * 16 + i;
    y[base + i] = vals[i];
    hbf[base + i] = f2bf((vals[i] - mu) * rstd * w[c] + bb[c]);
  }
}

// ---------------------------------------------------------------------------
// Launcher
// ---------------------------------------------------------------------------
extern "C" void kernel_launch(void* const* d_in, const int* in_sizes, int n_in,
                              void* d_out, int out_size, void* d_ws, size_t ws_size,
                              hipStream_t stream) {
  (void)in_sizes; (void)n_in; (void)out_size; (void)ws_size;
  const float* visual = (const float*)d_in[0];
  const float* text   = (const float*)d_in[1];
  const float* geo    = (const float*)d_in[2];
  const float* ln1w = (const float*)d_in[3];
  const float* ln1b = (const float*)d_in[4];
  const float* wq = (const float*)d_in[5];
  const float* bq = (const float*)d_in[6];
  const float* wk = (const float*)d_in[7];
  const float* bk = (const float*)d_in[8];
  const float* wvw = (const float*)d_in[9];
  const float* bv = (const float*)d_in[10];
  const float* wo = (const float*)d_in[11];
  const float* bo = (const float*)d_in[12];
  const float* gw = (const float*)d_in[13];
  const float* gb = (const float*)d_in[14];
  const float* logit_scale = (const float*)d_in[15];
  const float* alpha = (const float*)d_in[16];
  const float* ln2w = (const float*)d_in[17];
  const float* ln2b = (const float*)d_in[18];
  const float* w1 = (const float*)d_in[19];
  const float* b1 = (const float*)d_in[20];
  const float* w2 = (const float*)d_in[21];
  const float* b2 = (const float*)d_in[22];
  float* out = (float*)d_out;

  char* ws = (char*)d_ws;
  const size_t MB = 1ull << 20;
  float*          xln   = (float*)(ws);                       // 64 MiB: LN1(x) f32 / later y
  unsigned short* xbf   = (unsigned short*)(ws + 64 * MB);    // 32 MiB: LN1(x) bf16 / later LN2(y) bf16
  float*          qf    = (float*)(ws + 96 * MB);             // 64 MiB: q f32 / later o-proj out
  float*          gatef = (float*)(ws + 160 * MB);            // 64 MiB: gate f32
  unsigned short* gbf   = (unsigned short*)(ws + 96 * MB);    // 128 MiB: FFN mid bf16 (reuses qf+gatef)
  float*          kf    = (float*)(ws + 224 * MB);            // 1.3 MiB
  float*          vf    = (float*)(ws + 226 * MB);            // 1.3 MiB
  float*          pmk   = (float*)(ws + 228 * MB);            // pad mask
  float*          conf  = (float*)(ws + 229 * MB);            // conf_scale per row
  unsigned short* wqbf  = (unsigned short*)(ws + 230 * MB);
  unsigned short* wobf  = (unsigned short*)(ws + 231 * MB);
  unsigned short* gwbf  = (unsigned short*)(ws + 232 * MB);
  unsigned short* w1bf  = (unsigned short*)(ws + 233 * MB);   // 2 MiB
  unsigned short* w2bf  = (unsigned short*)(ws + 235 * MB);   // 2 MiB
  unsigned short* alnbf = (unsigned short*)(ws + 237 * MB);   // 32 MiB: aligned bf16
  float*          of    = qf;                                 // o-proj output reuses q buffer

  int wsz = CCH * CCH, w1sz = CCH * FFN;
  cvt_bf16_kernel<<<(wsz + 255) / 256, 256, 0, stream>>>(wq, wqbf, wsz);
  cvt_bf16_kernel<<<(wsz + 255) / 256, 256, 0, stream>>>(wo, wobf, wsz);
  cvt_bf16_kernel<<<(wsz + 255) / 256, 256, 0, stream>>>(gw, gwbf, wsz);
  cvt_bf16_kernel<<<(w1sz + 255) / 256, 256, 0, stream>>>(w1, w1bf, w1sz);
  cvt_bf16_kernel<<<(w1sz + 255) / 256, 256, 0, stream>>>(w2, w2bf, w1sz);

  ln1_kernel<<<NROW / 8, 256, 0, stream>>>(visual, ln1w, ln1b, xln, xbf);
  kv_proj_kernel<<<B_ * TT, 256, 0, stream>>>(text, wk, bk, wvw, bv, kf, vf);
  padmask_kernel<<<(B_ * TT + 7) / 8, 256, 0, stream>>>(text, pmk, B_ * TT);
  l2norm_kernel<<<(B_ * TT + 7) / 8, 256, 0, stream>>>(kf, B_ * TT);

  dim3 g512(NROW / 128, CCH / 64);
  gemm_bf16_kernel<0><<<g512, 256, 0, stream>>>(xbf, wqbf, bq, nullptr, nullptr,
                                                qf, nullptr, NROW, CCH, CCH);
  l2norm_kernel<<<NROW / 8, 256, 0, stream>>>(qf, NROW);

  attn_kernel<<<NROW / 4, 128, 0, stream>>>(qf, kf, vf, pmk, geo, logit_scale,
                                            alnbf, conf);

  gemm_bf16_kernel<1><<<g512, 256, 0, stream>>>(alnbf, wobf, bo, conf, nullptr,
                                                of, nullptr, NROW, CCH, CCH);
  gemm_bf16_kernel<2><<<g512, 256, 0, stream>>>(xbf, gwbf, gb, geo, nullptr,
                                                gatef, nullptr, NROW, CCH, CCH);

  combine_ln2_kernel<<<NROW / 8, 256, 0, stream>>>(xln, of, gatef, alpha,
                                                   ln2w, ln2b, xbf);

  dim3 gffn1(NROW / 128, FFN / 64);
  gemm_bf16_kernel<3><<<gffn1, 256, 0, stream>>>(xbf, w1bf, b1, nullptr, nullptr,
                                                 nullptr, gbf, NROW, FFN, CCH);
  gemm_bf16_kernel<4><<<g512, 256, 0, stream>>>(gbf, w2bf, b2, nullptr, xln,
                                                out, nullptr, NROW, CCH, FFN);
}
